// LogLikelihoodLoss_9569187136166
// MI455X (gfx1250) — matrix operand order repaired and checked by reading, not compiled, not run on hardware
//
#include <hip/hip_runtime.h>
#include <stdint.h>

// Problem dims (fixed by reference)
#define B_DIM 64
#define L_DIM 2048
#define H_DIM 256
#define K_DIM 32

typedef __attribute__((ext_vector_type(16))) __bf16 v16bf;
typedef __attribute__((ext_vector_type(8)))  float  v8f;

union FragBF {
    v16bf v;
    uint4 u[2];
};

// Fallback storage for the precomputed B fragments if d_ws is too small.
// Layout: frag f = chunk*2 + ntile, 32 lanes x 8 dwords (16 bf16) each -> 16 KB.
__device__ unsigned g_wfrag[16 * 32 * 8];

__device__ __forceinline__ unsigned short f2bf(float x) {
    unsigned u = __float_as_uint(x);
    unsigned r = (u + 0x7FFFu + ((u >> 16) & 1u)) >> 16;   // round-to-nearest-even
    return (unsigned short)r;
}

__device__ __forceinline__ unsigned pack2bf(float a, float b) {
#if __has_builtin(__builtin_amdgcn_cvt_pk_bf16_f32)
    typedef __attribute__((ext_vector_type(2))) __bf16 v2bf;
    v2bf r = __builtin_amdgcn_cvt_pk_bf16_f32(a, b);
    return __builtin_bit_cast(unsigned, r);
#else
    return (unsigned)f2bf(a) | ((unsigned)f2bf(b) << 16);
#endif
}

__device__ __forceinline__ float fast_rcp(float x) {
#if __has_builtin(__builtin_amdgcn_rcpf)
    return __builtin_amdgcn_rcpf(x);
#else
    return 1.0f / x;
#endif
}

__device__ __forceinline__ float fast_tanh(float x) {
#if __has_builtin(__builtin_amdgcn_tanhf)
    return __builtin_amdgcn_tanhf(x);         // v_tanh_f32 (gfx1250 TRANS op)
#else
    float t = __expf(2.0f * x);               // v_exp_f32
    return 1.0f - 2.0f * fast_rcp(t + 1.0f);  // v_rcp_f32
#endif
}

__device__ __forceinline__ float softplusf(float x) {
    return (x > 20.f) ? x : __logf(1.0f + __expf(x));
}

// One wave: zero the output scalar and bake W_int into WMMA B-fragment layout.
__global__ __launch_bounds__(32)
void nhp_prep(const float* __restrict__ W_int, unsigned* __restrict__ wfrag,
              float* __restrict__ out) {
    const int lane = threadIdx.x;
    if (lane == 0) out[0] = 0.f;
    const int grp = lane >> 4;     // 0/1 -> K half within chunk
    const int m   = lane & 15;     // N within tile
    for (int c = 0; c < H_DIM / 32; ++c) {
        for (int n = 0; n < 2; ++n) {
            const int j = n * 16 + m;                // output column
            unsigned frag[8];
            #pragma unroll
            for (int q = 0; q < 8; ++q) {
                int h0 = c * 32 + grp * 8 + 2 * q + (q >= 4 ? 8 : 0);  // reduction index (row of W)
                frag[q] = pack2bf(W_int[h0 * K_DIM + j], W_int[(h0 + 1) * K_DIM + j]);
            }
            unsigned* dst = &wfrag[(c * 2 + n) * 256 + lane * 8];
            *(uint4*)(dst)     = *(uint4*)(frag);
            *(uint4*)(dst + 4) = *(uint4*)(frag + 4);
        }
    }
}

__global__ __launch_bounds__(32)
void nhp_loss_kernel(const int*   __restrict__ event_seqs,   // [B, L]
                     const int*   __restrict__ seqs_length,  // [B]
                     const float* __restrict__ total_time,   // [B]
                     const float* __restrict__ hidden_t,     // [L, B, H]
                     const float* __restrict__ cell_t,       // [L, B, H]
                     const float* __restrict__ cell_target,  // [L, B, H]
                     const float* __restrict__ output_t,     // [L, B, H]
                     const float* __restrict__ decay_t,      // [L, B, H]
                     const float* __restrict__ sim_time,     // [L, B]
                     const float* __restrict__ b_int,        // [K]
                     const uint4* __restrict__ wfrag,        // precomputed B frags (kernarg!)
                     float*       __restrict__ out)
{
    constexpr int ROWPAD = H_DIM + 8;                 // bf16 elements per LDS row (bank pad)
    __shared__ unsigned short lds_sim[16 * ROWPAD];   // h_sim tile, bf16
    __shared__ unsigned short lds_hid[16 * ROWPAD];   // hidden tile, bf16

    const int lane = threadIdx.x;            // 0..31 (wave32)
    const int tile = blockIdx.x;             // 0..(L*B/16 - 1)
    const int r0   = tile * 16;              // first global row (row = l*B + b)
    const int l    = r0 / B_DIM;             // uniform across tile (16 | B)
    const int b0   = r0 % B_DIM;

    // ---- elementwise phase: decayed cell -> h_sim, plus hidden, to bf16 LDS tiles ----
    const float4* cell4 = (const float4*)cell_t;
    const float4* ctar4 = (const float4*)cell_target;
    const float4* outp4 = (const float4*)output_t;
    const float4* dec4  = (const float4*)decay_t;
    const float4* hid4  = (const float4*)hidden_t;
    const int base4 = r0 * (H_DIM / 4);      // float4 index of tile start
    #pragma unroll 4
    for (int i = 0; i < 32; ++i) {
        int idx  = i * 32 + lane;            // 0..1023 float4 within tile
        int row  = idx >> 6;                 // 64 float4 per row
        int col4 = idx & 63;
        float st = sim_time[r0 + row];
        float4 c  = cell4[base4 + idx];
        float4 ct = ctar4[base4 + idx];
        float4 o  = outp4[base4 + idx];
        float4 d  = dec4 [base4 + idx];
        float hx = o.x * fast_tanh(ct.x + (c.x - ct.x) * __expf(-d.x * st));
        float hy = o.y * fast_tanh(ct.y + (c.y - ct.y) * __expf(-d.y * st));
        float hz = o.z * fast_tanh(ct.z + (c.z - ct.z) * __expf(-d.z * st));
        float hw = o.w * fast_tanh(ct.w + (c.w - ct.w) * __expf(-d.w * st));
        unsigned* ps = (unsigned*)&lds_sim[row * ROWPAD + col4 * 4];
        ps[0] = pack2bf(hx, hy);
        ps[1] = pack2bf(hz, hw);
        float4 hv = hid4[base4 + idx];
        unsigned* ph = (unsigned*)&lds_hid[row * ROWPAD + col4 * 4];
        ph[0] = pack2bf(hv.x, hv.y);
        ph[1] = pack2bf(hv.z, hv.w);
    }
    __syncthreads();

    // ---- WMMA phase: [16 x 256] @ [256 x 32], two N-tiles, 8 K-chunks of 32 ----
    const int grp = lane >> 4;               // 0/1 : selects K/M halves per ISA layout
    const int m   = lane & 15;
    v8f acc_s0 = {}; v8f acc_s1 = {};        // sim logits, cols 0-15 / 16-31
    v8f acc_h0 = {}; v8f acc_h1 = {};        // hidden logits
    const uint4* wf = wfrag + lane * 2;      // each fragment: 64 uint4 apart
    #pragma unroll
    for (int c = 0; c < H_DIM / 32; ++c) {
        FragBF As, Ah, B0, B1;
        // A 16x32 bf16: lane m holds row m; K = c*32 + {grp*8..+7} then {16+grp*8..+7}
        const unsigned short* sa = &lds_sim[m * ROWPAD + c * 32 + grp * 8];
        As.u[0] = *(const uint4*)(sa);
        As.u[1] = *(const uint4*)(sa + 16);
        const unsigned short* ha = &lds_hid[m * ROWPAD + c * 32 + grp * 8];
        Ah.u[0] = *(const uint4*)(ha);
        Ah.u[1] = *(const uint4*)(ha + 16);
        // B fragments: precomputed, coalesced 32B/lane, hot in L2
        B0.u[0] = wf[(c * 2 + 0) * 64];
        B0.u[1] = wf[(c * 2 + 0) * 64 + 1];
        B1.u[0] = wf[(c * 2 + 1) * 64];
        B1.u[1] = wf[(c * 2 + 1) * 64 + 1];

        acc_s0 = __builtin_amdgcn_wmma_f32_16x16x32_bf16(false, As.v, false, B0.v,
                                                         (short)0, acc_s0, false, false);
        acc_s1 = __builtin_amdgcn_wmma_f32_16x16x32_bf16(false, As.v, false, B1.v,
                                                         (short)0, acc_s1, false, false);
        acc_h0 = __builtin_amdgcn_wmma_f32_16x16x32_bf16(false, Ah.v, false, B0.v,
                                                         (short)0, acc_h0, false, false);
        acc_h1 = __builtin_amdgcn_wmma_f32_16x16x32_bf16(false, Ah.v, false, B1.v,
                                                         (short)0, acc_h1, false, false);
    }

    // ---- epilogue: softplus sums (term2), event-column gather (term1), masked ----
    const float bias_lo = b_int[m];          // column m      (acc*0)
    const float bias_hi = b_int[16 + m];     // column 16+m   (acc*1)
    float partial = 0.f;
    #pragma unroll
    for (int v = 0; v < 8; ++v) {
        int M = grp * 8 + v;                 // C layout: lanes 0-15 rows 0-7, 16-31 rows 8-15
        int b = b0 + M;
        int len = seqs_length[b];
        bool valid = (l < len);
        // term 2: Monte-Carlo integral contribution (mask + total/len coefficient)
        float coef = valid ? (total_time[b] * fast_rcp((float)len)) : 0.f;
        float s = softplusf(acc_s0[v] + bias_lo) + softplusf(acc_s1[v] + bias_hi);
        partial += coef * s;
        // term 1: log-intensity at the next observed event type
        if (valid) {
            int lp1 = (l + 1 < L_DIM) ? (l + 1) : 0;
            int k = event_seqs[b * L_DIM + lp1];
            if ((k & 15) == m) {             // exactly one lane owns column k of this row
                float a = (k < 16) ? (acc_h0[v] + bias_lo) : (acc_h1[v] + bias_hi);
                partial -= __logf(softplusf(a));
            }
        }
    }
    // wave32 reduction, then one atomic per wave. loss = simulated_ll - original_ll
    #pragma unroll
    for (int off = 16; off > 0; off >>= 1)
        partial += __shfl_xor(partial, off, 32);
    if (lane == 0) atomicAdd(out, partial);
}

extern "C" void kernel_launch(void* const* d_in, const int* in_sizes, int n_in,
                              void* d_out, int out_size, void* d_ws, size_t ws_size,
                              hipStream_t stream) {
    (void)in_sizes; (void)n_in; (void)out_size;
    const int*   event_seqs  = (const int*)  d_in[0];
    const int*   seqs_length = (const int*)  d_in[1];
    const float* total_time  = (const float*)d_in[2];
    const float* hidden_t    = (const float*)d_in[3];
    const float* cell_t      = (const float*)d_in[4];
    const float* cell_target = (const float*)d_in[5];
    const float* output_t    = (const float*)d_in[6];
    const float* decay_t     = (const float*)d_in[7];
    const float* sim_time    = (const float*)d_in[8];
    const float* W_int       = (const float*)d_in[9];
    const float* b_int       = (const float*)d_in[10];
    float* out = (float*)d_out;

    // B-fragment buffer: prefer harness scratch; fall back to device global.
    // (hipGetSymbolAddress is a host-side query: no stream ops, graph-capture safe,
    //  called only when d_ws is null or under 16 KB.)
    void* wbuf = d_ws;
    if (ws_size < 16u * 1024u || wbuf == nullptr) {
        (void)hipGetSymbolAddress(&wbuf, HIP_SYMBOL(g_wfrag));
    }

    nhp_prep<<<1, 32, 0, stream>>>(W_int, (unsigned*)wbuf, out);
    const int ntiles = (L_DIM * B_DIM) / 16;   // 8192 tiles of 16 (l,b)-rows
    nhp_loss_kernel<<<ntiles, 32, 0, stream>>>(event_seqs, seqs_length, total_time,
                                               hidden_t, cell_t, cell_target, output_t,
                                               decay_t, sim_time, b_int,
                                               (const uint4*)wbuf, out);
}
